// Aligner_37752762532605
// MI455X (gfx1250) — compile-verified, matrix operand order checked by reading, not run on hardware
//
#include <hip/hip_runtime.h>

// ---------------------------------------------------------------------------
// Types & WMMA wrapper (fp32 16x16x4 confirmed lowering on gfx1250)
// ---------------------------------------------------------------------------
typedef __attribute__((ext_vector_type(2))) float v2f;
typedef __attribute__((ext_vector_type(8))) float v8f;

#if __has_builtin(__builtin_amdgcn_wmma_f32_16x16x4_f32)
__device__ __forceinline__ v8f wmma4(v2f a, v2f b, v8f c) {
  // D = A(16x4) * B(4x16) + C, all fp32
  return __builtin_amdgcn_wmma_f32_16x16x4_f32(false, a, false, b, (short)0, c,
                                               false, false);
}
#else
typedef __attribute__((ext_vector_type(16))) _Float16 v16h;
__device__ __forceinline__ v8f wmma4(v2f a, v2f b, v8f c) {
  v16h ah = {0,0,0,0,0,0,0,0,0,0,0,0,0,0,0,0};
  v16h bh = {0,0,0,0,0,0,0,0,0,0,0,0,0,0,0,0};
  ah[0] = (_Float16)a.x; ah[1] = (_Float16)a.y;
  bh[0] = (_Float16)b.x; bh[1] = (_Float16)b.y;
  return __builtin_amdgcn_wmma_f32_16x16x32_f16(false, ah, false, bh, (short)0, c,
                                                false, false);
}
#endif

// ---------------------------------------------------------------------------
// Async global->LDS copy (ASYNCcnt-tracked) with safe fallback.
// Probe feedback: param 0 is AS1 (global) int*, so use int-typed AS pointers.
// ---------------------------------------------------------------------------
typedef __attribute__((address_space(1))) int gint;
typedef __attribute__((address_space(3))) int lint;

#if __has_builtin(__builtin_amdgcn_global_load_async_to_lds_b32)
#define HAVE_ASYNC_LDS 1
__device__ __forceinline__ void async_g2l_b32(void* lds, const void* g) {
  __builtin_amdgcn_global_load_async_to_lds_b32((gint*)(void*)g, (lint*)lds, 0, 0);
}
#else
#define HAVE_ASYNC_LDS 0
#endif

__device__ __forceinline__ void wait_async_all() {
#if __has_builtin(__builtin_amdgcn_s_wait_asynccnt)
  __builtin_amdgcn_s_wait_asynccnt(0);
#else
  asm volatile("s_wait_asynccnt 0x0" ::: "memory");
#endif
}

#define DD   128
#define WW   65
#define WP   80      // padded rfft width (5 x 16 tiles)
#define NB   8
#define NK   125
#define NBK  1000
#define MWIN 65      // 2*32+1 shift window
#define NEG_INF (-3.402823e38f)

// ---------------------------------------------------------------------------
// K0: twiddle tables (all fftshifts folded in) + ZYZ rotation matrices
// ---------------------------------------------------------------------------
__global__ void k_precompute(const float* __restrict__ euler,
                             float* TxR, float* TxI,   // 128 x 80 : forward rfft along x
                             float* WyR, float* WyI,   // 128 x 128: forward DFT, shifted rows
                             float* WyiR, float* WyiI, // 80 x 128 : inverse-y DFT (windowed out)
                             float* XR, float* XI,     // 65 x 80  : Hermitian x-inverse (windowed)
                             float* ROT)               // 125 x 9
{
  const int g = blockIdx.x * blockDim.x + threadIdx.x;
  const int stride = gridDim.x * blockDim.x;
  const float TWO_PI = 6.28318530717958647692f;

  for (int i = g; i < DD * WP; i += stride) {       // Tx: e^{-2*pi*i*n*k/128}
    int n = i / WP, k = i % WP;
    if (k < WW) {
      int m = (n * k) & 127;
      float t = TWO_PI * (float)m / 128.f;
      TxR[i] = __cosf(t); TxI[i] = -__sinf(t);
    } else { TxR[i] = 0.f; TxI[i] = 0.f; }
  }
  for (int i = g; i < DD * DD; i += stride) {       // Wy: shifted-output forward DFT
    int r = i / DD, y = i % DD;
    int f = r - 64;
    int m = ((f * y) % 128 + 128) % 128;
    float t = TWO_PI * (float)m / 128.f;
    WyR[i] = __cosf(t); WyI[i] = -__sinf(t);
  }
  for (int i = g; i < WP * DD; i += stride) {       // Wyi: inverse-y, output y in [32,96]
    int r = i / DD, r0 = i % DD;
    if (r < MWIN) {
      int f = r0 - 64;
      int m = ((f * (r - 32)) % 128 + 128) % 128;
      float t = TWO_PI * (float)m / 128.f;
      WyiR[i] = __cosf(t) * (1.f / 128.f);
      WyiI[i] = __sinf(t) * (1.f / 128.f);
    } else { WyiR[i] = 0.f; WyiI[i] = 0.f; }
  }
  for (int i = g; i < WW * WP; i += stride) {       // X: Hermitian irfft along x, windowed
    int kx = i / WP, wx = i % WP;
    if (wx < MWIN) {
      float s = (kx == 0 || kx == 64) ? 1.f : 2.f;
      int m = ((kx * (wx - 32)) % 128 + 128) % 128;
      float t = TWO_PI * (float)m / 128.f;
      XR[i] = s * __cosf(t) * (1.f / 128.f);
      XI[i] = -s * __sinf(t) * (1.f / 128.f);
    } else { XR[i] = 0.f; XI[i] = 0.f; }
  }
  for (int p = g; p < NK; p += stride) {            // R = Rz(a) Ry(b) Rz(c)
    const float d2r = 0.01745329251994329577f;
    float a = euler[p * 3 + 0] * d2r;
    float b = euler[p * 3 + 1] * d2r;
    float c = euler[p * 3 + 2] * d2r;
    float ca = __cosf(a), sa = __sinf(a);
    float cb = __cosf(b), sb = __sinf(b);
    float cc = __cosf(c), sc = __sinf(c);
    float* R = ROT + p * 9;
    R[0] = ca * cb * cc - sa * sc; R[1] = -ca * cb * sc - sa * cc; R[2] = ca * sb;
    R[3] = sa * cb * cc + ca * sc; R[4] = -sa * cb * sc + ca * cc; R[5] = sa * sb;
    R[6] = -sb * cc;               R[7] = sb * sc;                 R[8] = cb;
  }
}

// ---------------------------------------------------------------------------
// K1: real GEMM against rfft twiddles: C{re,im}[M x 80] = A[M x 128] * Tx
// one wave per 16x16 output tile (WMMA f32 16x16x4); A streams from HBM once.
// ---------------------------------------------------------------------------
__global__ void k_rgemm_rfft(const float* __restrict__ A,
                             const float* __restrict__ TxR,
                             const float* __restrict__ TxI,
                             float* __restrict__ CR, float* __restrict__ CI,
                             int Mtiles) {
  const int wave = (blockIdx.x * blockDim.x + threadIdx.x) >> 5;
  const int lane = threadIdx.x & 31;
  const int nt = wave % 5;
  const int mt = wave / 5;
  if (mt >= Mtiles) return;
  const int l16 = lane & 15, lh = lane >> 4;

  v8f cre = {0.f,0.f,0.f,0.f,0.f,0.f,0.f,0.f};
  v8f cim = {0.f,0.f,0.f,0.f,0.f,0.f,0.f,0.f};
  const int arow = (mt * 16 + l16) * DD;
  for (int k0 = 0; k0 < DD; k0 += 4) {
    v2f a; a.x = A[arow + k0 + 2 * lh]; a.y = A[arow + k0 + 2 * lh + 1];
    const int b0 = (k0 + 2 * lh) * WP + nt * 16 + l16;
    v2f br; br.x = TxR[b0]; br.y = TxR[b0 + WP];
    v2f bi; bi.x = TxI[b0]; bi.y = TxI[b0 + WP];
    cre = wmma4(a, br, cre);
    cim = wmma4(a, bi, cim);
    if (k0 + 4 < DD) __builtin_prefetch(&A[arow + k0 + 4], 0, 0);
  }
  const int crow = (mt * 16 + lh * 8) * WP + nt * 16 + l16;
#pragma unroll
  for (int j = 0; j < 8; ++j) { CR[crow + j * WP] = cre[j]; CI[crow + j * WP] = cim[j]; }
}

// ---------------------------------------------------------------------------
// K2: batched complex GEMM with LDS-staged B strip:
//   C[batch][:, strip*80 .. +80] = W(complex 128x128) * B[batch][:, strip*80 ..]
// block = 256 (8 waves); B strip (128x80 complex, 80 KB) loaded to LDS via
// async global->LDS; wave w owns m-tile w and sweeps 5 n-tiles (A reuse x5,
// B reuse x8 across waves).
// ---------------------------------------------------------------------------
__global__ void k_cgemm_lds(const float* __restrict__ WR, const float* __restrict__ WI,
                            const float* __restrict__ BR, const float* __restrict__ BI,
                            float* __restrict__ CR, float* __restrict__ CI,
                            int Ncols, long bStrideB, long bStrideC, int strips) {
  extern __shared__ float smem[];
  float* sBR = smem;
  float* sBI = smem + DD * WP;

  const int batch = blockIdx.x / strips;
  const int strip = blockIdx.x % strips;
  const int tid = threadIdx.x;

  const float* bR = BR + (long)batch * bStrideB + (long)strip * WP;
  const float* bI = BI + (long)batch * bStrideB + (long)strip * WP;

  for (int i = tid; i < DD * WP; i += 256) {
    const long off = (long)(i / WP) * Ncols + (i % WP);
#if HAVE_ASYNC_LDS
    async_g2l_b32(&sBR[i], &bR[off]);
    async_g2l_b32(&sBI[i], &bI[off]);
#else
    sBR[i] = bR[off];
    sBI[i] = bI[off];
#endif
  }
#if HAVE_ASYNC_LDS
  wait_async_all();
#endif
  __syncthreads();

  const int wave = tid >> 5, lane = tid & 31;
  const int l16 = lane & 15, lh = lane >> 4;
  const v8f z8 = {0.f,0.f,0.f,0.f,0.f,0.f,0.f,0.f};
  v8f cre[5], cim[5];
#pragma unroll
  for (int nt = 0; nt < 5; ++nt) { cre[nt] = z8; cim[nt] = z8; }

  const int arow = (wave * 16 + l16) * DD;
  for (int k0 = 0; k0 < DD; k0 += 4) {
    const int ao = arow + k0 + 2 * lh;
    v2f ar; ar.x = WR[ao]; ar.y = WR[ao + 1];
    v2f ai; ai.x = WI[ao]; ai.y = WI[ao + 1];
    v2f nai = -ai;
    const int bbase = (k0 + 2 * lh) * WP + l16;
#pragma unroll
    for (int nt = 0; nt < 5; ++nt) {
      const int bo = bbase + nt * 16;
      v2f br; br.x = sBR[bo]; br.y = sBR[bo + WP];
      v2f bi; bi.x = sBI[bo]; bi.y = sBI[bo + WP];
      cre[nt] = wmma4(ar, br, cre[nt]);
      cre[nt] = wmma4(nai, bi, cre[nt]);
      cim[nt] = wmma4(ar, bi, cim[nt]);
      cim[nt] = wmma4(ai, br, cim[nt]);
    }
  }
  const long cbase = (long)batch * bStrideC + (long)strip * WP;
#pragma unroll
  for (int nt = 0; nt < 5; ++nt) {
    v8f vr = cre[nt], vi = cim[nt];
#pragma unroll
    for (int j = 0; j < 8; ++j) {
      const int row = wave * 16 + lh * 8 + j;
      CR[cbase + (long)row * Ncols + nt * 16 + l16] = vr[j];
      CI[cbase + (long)row * Ncols + nt * 16 + l16] = vi[j];
    }
  }
}

// ---------------------------------------------------------------------------
// K3: central-slice extraction (Fourier gather + trilinear + Hermitian conj)
// ---------------------------------------------------------------------------
__global__ void k_project(const float* __restrict__ VfR, const float* __restrict__ VfI,
                          const float* __restrict__ ROT,
                          float* __restrict__ PjR, float* __restrict__ PjI) {
  const int idx = blockIdx.x * blockDim.x + threadIdx.x;
  if (idx >= NK * DD * WP) return;
  const int c = idx % WP;
  const int r = (idx / WP) % DD;
  const int p = idx / (DD * WP);
  if (c >= WW) { PjR[idx] = 0.f; PjI[idx] = 0.f; return; }

  const float x = (float)c, y = (float)(r - 64);
  const float* R = ROT + p * 9;
  float rx = R[0] * x + R[1] * y;
  float ry = R[3] * x + R[4] * y;
  float rz = R[6] * x + R[7] * y;
  const bool cj = rx < 0.f;
  if (cj) { rx = -rx; ry = -ry; rz = -rz; }
  const float xi = rx, yi = ry + 64.f, zi = rz + 64.f;

  const float z0f = floorf(zi), y0f = floorf(yi), x0f = floorf(xi);
  const int z0 = (int)z0f, y0 = (int)y0f, x0 = (int)x0f;
  const float tz = zi - z0f, ty = yi - y0f, tx = xi - x0f;

  float sr = 0.f, si = 0.f;
#pragma unroll
  for (int dz = 0; dz < 2; ++dz) {
    const int zc = z0 + dz;
    const float wz = dz ? tz : 1.f - tz;
    const bool vz = (zc >= 0) && (zc < DD);
    const int zx = zc < 0 ? 0 : (zc > DD - 1 ? DD - 1 : zc);
#pragma unroll
    for (int dy = 0; dy < 2; ++dy) {
      const int yc = y0 + dy;
      const float wy = dy ? ty : 1.f - ty;
      const bool vy = (yc >= 0) && (yc < DD);
      const int yx = yc < 0 ? 0 : (yc > DD - 1 ? DD - 1 : yc);
#pragma unroll
      for (int dx = 0; dx < 2; ++dx) {
        const int xc = x0 + dx;
        const float wx = dx ? tx : 1.f - tx;
        const bool vx = (xc >= 0) && (xc < WW);
        const int xx = xc < 0 ? 0 : (xc > WW - 1 ? WW - 1 : xc);
        const float w = (vz && vy && vx) ? wz * wy * wx : 0.f;
        const int off = (zx * DD + yx) * WP + xx;
        sr += w * VfR[off];
        si += w * VfI[off];
      }
    }
  }
  if (cj) si = -si;
  if (x * x + y * y > 4096.f) { sr = 0.f; si = 0.f; }
  PjR[idx] = sr; PjI[idx] = si;
}

// ---------------------------------------------------------------------------
// K4: fused correlation GEMM. Per (particle b, pose p):
//   prod = Pf[b] * ctf[b] * conj(Pj[p])  (computed into LDS, never hits HBM)
//   Ciy  = Wyi(80x128 complex) x prod(128x80 complex)     via WMMA
// blockDim = 160 (5 waves, one m-tile row each, 5 n-tiles per wave)
// ---------------------------------------------------------------------------
__global__ void k_corr_gemm(const float* __restrict__ WyiR, const float* __restrict__ WyiI,
                            const float* __restrict__ PfR, const float* __restrict__ PfI,
                            const float* __restrict__ PjR, const float* __restrict__ PjI,
                            const float* __restrict__ ctf,
                            float* __restrict__ CyR, float* __restrict__ CyI) {
  extern __shared__ float smem[];
  float* sPR = smem;
  float* sPI = smem + DD * WP;

  const int bk = blockIdx.x;
  const int b = bk / NK, p = bk % NK;
  const int tid = threadIdx.x;

  const float* fR = PfR + (long)b * DD * WP;
  const float* fI = PfI + (long)b * DD * WP;
  const float* gR = PjR + (long)p * DD * WP;
  const float* gI = PjI + (long)p * DD * WP;

  for (int i = tid; i < DD * WP; i += 160) {
    const int rr = i / WP, cc = i % WP;
    const float ar = fR[i], ai = fI[i];
    const float pr = gR[i], pi = gI[i];
    const float cf = (cc < WW) ? ctf[(b * DD + rr) * WW + cc] : 0.f;
    sPR[i] = cf * (ar * pr + ai * pi);   // Re( Pf * conj(Pj) ) * ctf
    sPI[i] = cf * (ai * pr - ar * pi);   // Im( Pf * conj(Pj) ) * ctf
  }
  __syncthreads();

  const int wave = tid >> 5, lane = tid & 31;
  const int l16 = lane & 15, lh = lane >> 4;
  const v8f z8 = {0.f,0.f,0.f,0.f,0.f,0.f,0.f,0.f};
  v8f cre[5], cim[5];
#pragma unroll
  for (int nt = 0; nt < 5; ++nt) { cre[nt] = z8; cim[nt] = z8; }

  const int arow = (wave * 16 + l16) * DD;
  for (int k0 = 0; k0 < DD; k0 += 4) {
    const int ao = arow + k0 + 2 * lh;
    v2f ar2; ar2.x = WyiR[ao]; ar2.y = WyiR[ao + 1];
    v2f ai2; ai2.x = WyiI[ao]; ai2.y = WyiI[ao + 1];
    v2f nai = -ai2;
    const int bbase = (k0 + 2 * lh) * WP + l16;
#pragma unroll
    for (int nt = 0; nt < 5; ++nt) {
      const int bo = bbase + nt * 16;
      v2f br; br.x = sPR[bo]; br.y = sPR[bo + WP];
      v2f bi; bi.x = sPI[bo]; bi.y = sPI[bo + WP];
      cre[nt] = wmma4(ar2, br, cre[nt]);
      cre[nt] = wmma4(nai, bi, cre[nt]);
      cim[nt] = wmma4(ar2, bi, cim[nt]);
      cim[nt] = wmma4(ai2, br, cim[nt]);
    }
  }
  const long obase = (long)bk * (WP * WP);
#pragma unroll
  for (int nt = 0; nt < 5; ++nt) {
    v8f vr = cre[nt], vi = cim[nt];
#pragma unroll
    for (int j = 0; j < 8; ++j) {
      const int row = wave * 16 + lh * 8 + j;
      CyR[obase + row * WP + nt * 16 + l16] = vr[j];
      CyI[obase + row * WP + nt * 16 + l16] = vi[j];
    }
  }
}

// ---------------------------------------------------------------------------
// K5: Hermitian x-inverse restricted to 65x65 shift window + block top-2
// (Ciy image staged to LDS via async global->LDS copy)
// ---------------------------------------------------------------------------
__global__ void k_xinv_topk(const float* __restrict__ CyR, const float* __restrict__ CyI,
                            const float* __restrict__ XR, const float* __restrict__ XI,
                            float* __restrict__ out) {
  extern __shared__ float smem[];
  float* sR = smem;
  float* sI = smem + MWIN * WP;
  __shared__ float rv[512];
  __shared__ int ri[512];

  const int bk = blockIdx.x, tid = threadIdx.x;
  const float* cR = CyR + (long)bk * (WP * WP);
  const float* cI = CyI + (long)bk * (WP * WP);
  for (int i = tid; i < MWIN * WP; i += 256) {
#if HAVE_ASYNC_LDS
    async_g2l_b32(&sR[i], &cR[i]);
    async_g2l_b32(&sI[i], &cI[i]);
#else
    sR[i] = cR[i];
    sI[i] = cI[i];
#endif
  }
#if HAVE_ASYNC_LDS
  wait_async_all();
#endif
  __syncthreads();

  float v1 = NEG_INF, v2 = NEG_INF;
  int i1 = 0, i2 = 0;
  for (int o = tid; o < MWIN * MWIN; o += 256) {
    const int wy = o / MWIN, wx = o % MWIN;
    const float* rrow = sR + wy * WP;
    const float* irow = sI + wy * WP;
    float acc = 0.f;
    for (int kx = 0; kx < WW; ++kx)
      acc += rrow[kx] * XR[kx * WP + wx] + irow[kx] * XI[kx * WP + wx];
    const int fi = (wy + 32) * DD + (wx + 32);
    if (acc > v1)      { v2 = v1; i2 = i1; v1 = acc; i1 = fi; }
    else if (acc > v2) { v2 = acc; i2 = fi; }
  }
  rv[tid * 2] = v1; ri[tid * 2] = i1;
  rv[tid * 2 + 1] = v2; ri[tid * 2 + 1] = i2;
  __syncthreads();
  if (tid == 0) {
    float b1 = NEG_INF, b2 = NEG_INF;
    int j1 = 0, j2 = 0;
    for (int s = 0; s < 512; ++s) {
      const float v = rv[s];
      if (v > b1)      { b2 = b1; j2 = j1; b1 = v; j1 = ri[s]; }
      else if (v > b2) { b2 = v; j2 = ri[s]; }
    }
    out[bk * 2] = b1;
    out[bk * 2 + 1] = b2;
    float* sh = out + 2 * NBK;                       // shifts after all vals
    sh[(bk * 2) * 2]     = (float)(j1 / DD - 64);
    sh[(bk * 2) * 2 + 1] = (float)(j1 % DD - 64);
    sh[(bk * 2 + 1) * 2]     = (float)(j2 / DD - 64);
    sh[(bk * 2 + 1) * 2 + 1] = (float)(j2 % DD - 64);
  }
}

// ---------------------------------------------------------------------------
// Host orchestration
// ---------------------------------------------------------------------------
extern "C" void kernel_launch(void* const* d_in, const int* in_sizes, int n_in,
                              void* d_out, int out_size, void* d_ws, size_t ws_size,
                              hipStream_t stream) {
  (void)in_sizes; (void)n_in; (void)out_size; (void)ws_size;
  const float* vol   = (const float*)d_in[0];
  const float* parts = (const float*)d_in[1];
  const float* ctf   = (const float*)d_in[2];
  const float* euler = (const float*)d_in[3];
  float* W = (float*)d_ws;

  size_t o = 0;
  float* TxR  = W + o; o += 128 * 80;
  float* TxI  = W + o; o += 128 * 80;
  float* WyR  = W + o; o += 128 * 128;
  float* WyI  = W + o; o += 128 * 128;
  float* WyiR = W + o; o += 80 * 128;
  float* WyiI = W + o; o += 80 * 128;
  float* XR   = W + o; o += 65 * 80;
  float* XI   = W + o; o += 65 * 80;
  float* ROT  = W + o; o += 1280;
  float* AxR  = W + o; o += 16384 * 80;   // ping buffer (re)
  float* AxI  = W + o; o += 16384 * 80;
  float* ByR  = W + o; o += 16384 * 80;   // pong buffer (re)
  float* ByI  = W + o; o += 16384 * 80;
  float* PxR  = W + o; o += 1024 * 80;
  float* PxI  = W + o; o += 1024 * 80;
  float* PfR  = W + o; o += 1024 * 80;
  float* PfI  = W + o; o += 1024 * 80;
  float* PjR  = W + o; o += 125 * 128 * 80;
  float* PjI  = W + o; o += 125 * 128 * 80;
  float* CyR  = W + o; o += 1000 * 80 * 80;
  float* CyI  = W + o; o += 1000 * 80 * 80;

  const size_t ldsB = 2 * 128 * 80 * sizeof(float);   // 80 KB B-strip

  // twiddles + rotmats
  k_precompute<<<64, 256, 0, stream>>>(euler, TxR, TxI, WyR, WyI, WyiR, WyiI, XR, XI, ROT);

  // volume 3D rfft: x-pass real GEMM, then shifted y / z complex GEMMs
  k_rgemm_rfft<<<640, 256, 0, stream>>>(vol, TxR, TxI, AxR, AxI, 1024);
  // y-pass: 128 z-slabs, one 80-col strip each
  k_cgemm_lds<<<128, 256, ldsB, stream>>>(WyR, WyI, AxR, AxI, ByR, ByI,
                                          80, 10240L, 10240L, 1);
  // z-pass: single batch, 128 strips across the 10240-wide (ky,kx) plane
  k_cgemm_lds<<<128, 256, ldsB, stream>>>(WyR, WyI, ByR, ByI, AxR, AxI,
                                          10240, 0L, 0L, 128);          // -> Vf in Ax

  // particles 2D rfft (x then shifted y)
  k_rgemm_rfft<<<40, 256, 0, stream>>>(parts, TxR, TxI, PxR, PxI, 64);
  k_cgemm_lds<<<8, 256, ldsB, stream>>>(WyR, WyI, PxR, PxI, PfR, PfI,
                                        80, 10240L, 10240L, 1);

  // central slice extraction for all 125 poses
  k_project<<<5000, 256, 0, stream>>>(AxR, AxI, ROT, PjR, PjI);

  // fused (parts * ctf * conj(proj)) -> inverse-y DFT, windowed to 65 output rows
  k_corr_gemm<<<1000, 160, ldsB, stream>>>(WyiR, WyiI, PfR, PfI, PjR, PjI, ctf,
                                           CyR, CyI);

  // Hermitian x-inverse over 65x65 window + top-2 peak extraction
  k_xinv_topk<<<1000, 256, 2 * 65 * 80 * sizeof(float), stream>>>(
      CyR, CyI, XR, XI, (float*)d_out);
}
